// Decoder_28741921145342
// MI455X (gfx1250) — matrix operand order repaired and checked by reading, not compiled, not run on hardware
//
#include <hip/hip_runtime.h>
#include <hip/hip_bf16.h>

typedef __attribute__((ext_vector_type(16))) _Float16 v16h;
typedef __attribute__((ext_vector_type(8)))  _Float16 v8h;
typedef __attribute__((ext_vector_type(8)))  float    v8f;

#define BSZ 1024
#define TM1 127
#define HD  128
#define HE  128

__device__ __forceinline__ float fast_tanh(float x) {
  x = fminf(10.0f, fmaxf(-10.0f, x));
  float e = __expf(2.0f * x);
  return (e - 1.0f) / (e + 1.0f);
}
__device__ __forceinline__ float fast_sigm(float x) {
  return 1.0f / (1.0f + __expf(-x));
}

// A-fragment for v_wmma_f32_16x16x32_f16: lane L holds row
// M=L&15; lanes 0-15 get K = ks..ks+15 with ks=kf*32,
// lanes 16-31 the same shifted by 16. 'row' points at element (M, 0) of a
// row-major f16 matrix; ks already includes the (lane>>4)*16 shift.
__device__ __forceinline__ v16h load_afrag(const _Float16* row, int ks) {
  v8h lo = *(const v8h*)(row + ks);
  v8h hi = *(const v8h*)(row + ks + 8);
  v16h r;
#pragma unroll
  for (int i = 0; i < 8; ++i) { r[i] = lo[i]; r[i + 8] = hi[i]; }
  return r;
}

// ---------------------------------------------------------------------------
// K1: transpose + f16-convert weights into workspace.
//   W1dcT[n][k] = W1[k][n]           n<128, k<256   (rows 0..255 of W1 = [W1_d;W1_c])
//   W1xT [n][k] = W1[256+k][n]       n<128, k<128
//   WhT  [n][k] = Wh[k][n]           n<512, k<128
// ---------------------------------------------------------------------------
__global__ void __launch_bounds__(256) prep_weights(
    const float* __restrict__ W1, const float* __restrict__ Wh,
    _Float16* __restrict__ W1dcT, _Float16* __restrict__ W1xT,
    _Float16* __restrict__ WhT) {
  int i = blockIdx.x * 256 + threadIdx.x;
  if (i < 32768) {
    int k = i & 255, n = i >> 8;
    W1dcT[n * 256 + k] = (_Float16)W1[k * 128 + n];
  } else if (i < 49152) {
    int j = i - 32768;
    int k = j & 127, n = j >> 7;
    W1xT[n * 128 + k] = (_Float16)W1[(256 + k) * 128 + n];
  } else if (i < 114688) {
    int j = i - 49152;
    int k = j & 127, n = j >> 7;
    WhT[n * 128 + k] = (_Float16)Wh[k * 512 + n];
  }
}

// K2: f16 copy of X_encoded (stays hot in 192MB L2; halves per-step traffic).
__global__ void __launch_bounds__(256) conv_x(const float* __restrict__ X,
                                              _Float16* __restrict__ Xh, int n) {
  int i = blockIdx.x * 256 + threadIdx.x;
  if (i < n) Xh[i] = (_Float16)X[i];
}

// ---------------------------------------------------------------------------
// K3: pre_x = X @ W1_x + b1, stored f16.  M = 1024*127 = 130048, K = N = 128.
// One wave per 16-row M-tile; 8 waves/block -> 1016 blocks.
// ---------------------------------------------------------------------------
__global__ void __launch_bounds__(256) prex_gemm(
    const _Float16* __restrict__ Xh, const _Float16* __restrict__ W1xT,
    const float* __restrict__ b1, _Float16* __restrict__ preXh) {
  const int lane = threadIdx.x & 31;
  const int w    = threadIdx.x >> 5;
  const int nn   = lane & 15;
  const int hi   = lane >> 4;
  const int mtile = blockIdx.x * 8 + w;

  const _Float16* arow = Xh + (size_t)(mtile * 16 + nn) * HE;
  v16h a[4];
#pragma unroll
  for (int kf = 0; kf < 4; ++kf) a[kf] = load_afrag(arow, kf * 32 + hi * 16);

#pragma unroll
  for (int nt = 0; nt < 8; ++nt) {
    v8f acc = {};
#pragma unroll
    for (int kf = 0; kf < 4; ++kf) {
      // B-fragment: lane holds column N = nn of tile nt, 16 contiguous K values.
      v16h b = *(const v16h*)(W1xT + (size_t)(nt * 16 + nn) * 128 + kf * 32 + hi * 16);
      acc = __builtin_amdgcn_wmma_f32_16x16x32_f16(false, a[kf], false, b,
                                                   (short)0, acc, false, false);
    }
    int col = nt * 16 + nn;
    float bb = b1[col];
#pragma unroll
    for (int r = 0; r < 8; ++r) {
      int m = r + hi * 8;  // C/D layout: lanes16-31 hold M = r+8
      preXh[(size_t)(mtile * 16 + m) * HE + col] = (_Float16)(acc[r] + bb);
    }
  }
}

// ---------------------------------------------------------------------------
// Main recurrent kernel: one block = 16 batch rows, 127 time steps.
// State lives in LDS; pre_x/X stream from L2 as f16 each step.
// ---------------------------------------------------------------------------
__global__ void __launch_bounds__(256) decoder_main(
    const float* __restrict__ X, const float* __restrict__ y_prev,
    const float* __restrict__ W2, const float* __restrict__ b2,
    const float* __restrict__ Wfc, const float* __restrict__ bfc,
    const float* __restrict__ Wx, const float* __restrict__ bl,
    const float* __restrict__ Wf, const float* __restrict__ bf,
    const _Float16* __restrict__ Xh, const _Float16* __restrict__ preXh,
    const _Float16* __restrict__ W1dcT, const _Float16* __restrict__ WhT,
    float* __restrict__ out) {
  __shared__ float    Us[16][128];     // U = d@W1_d + c@W1_c
  __shared__ float    Cs[16][128];     // cell state (f32)
  __shared__ float    Ds[16][128];     // hidden state (f32)
  __shared__ float    ctxs[16][128];   // attention context
  __shared__ float    betas[16][128];  // attention scores (127 used)
  __shared__ _Float16 DCh[16][256];    // [d | c] as f16: WMMA A operand
  __shared__ float    W2s[128];
  __shared__ float    Wfcs[129];
  __shared__ float    yts[16];

  const int tid  = threadIdx.x;
  const int lane = tid & 31;
  const int w    = tid >> 5;   // wave 0..7
  const int nn   = lane & 15;
  const int hi   = lane >> 4;
  const int bg0  = blockIdx.x * 16;
  const int ncol = w * 16 + nn;  // this lane's output column within 0..127

  if (tid < 128) W2s[tid] = W2[tid];
  if (tid < 129) Wfcs[tid] = Wfc[tid];

  // init: d0 = c0 = broadcast X_encoded[b,0,0]
  {
    int b = tid >> 4, c0 = (tid & 15) * 8;
    float v = X[(size_t)(bg0 + b) * (TM1 * HE)];
#pragma unroll
    for (int j = 0; j < 8; ++j) {
      Ds[b][c0 + j] = v;
      Cs[b][c0 + j] = v;
      DCh[b][c0 + j] = (_Float16)v;
      DCh[b][128 + c0 + j] = (_Float16)v;
    }
  }

  const float b2v  = b2[0];
  const float bfcv = bfc[0];
  // per-lane gate-column constants (hoisted out of the time loop)
  const float wxi = Wx[ncol], wxf = Wx[128 + ncol], wxg = Wx[256 + ncol], wxo = Wx[384 + ncol];
  const float bli = bl[ncol], blf = bl[128 + ncol], blg = bl[256 + ncol], blo = bl[384 + ncol];

  __syncthreads();

  for (int t = 0; t < TM1; ++t) {
    // --- Phase A: U(16,128) = [d|c](16,256) @ W1dc(256,128); wave w -> N-tile w
    {
      v8f acc = {};
#pragma unroll
      for (int kf = 0; kf < 8; ++kf) {
        v16h a = load_afrag(&DCh[nn][0], kf * 32 + hi * 16);
        v16h b = *(const v16h*)(W1dcT + (size_t)ncol * 256 + kf * 32 + hi * 16);
        acc = __builtin_amdgcn_wmma_f32_16x16x32_f16(false, a, false, b,
                                                     (short)0, acc, false, false);
      }
#pragma unroll
      for (int r = 0; r < 8; ++r) Us[r + hi * 8][ncol] = acc[r];
    }
    __syncthreads();

    // --- Phase B: beta[b][t'] = tanh(pre_x + U) . W2 + b2  (VALU/TRANS bound)
    {
      int b = tid >> 4, tb = tid & 15;
      const _Float16* px = preXh + (size_t)(bg0 + b) * (TM1 * HE);
#pragma unroll 1
      for (int i = 0; i < 8; ++i) {
        int tp = tb + 16 * i;
        if (tp < TM1) {
          float s = 0.0f;
          const _Float16* row = px + tp * HE;
#pragma unroll 4
          for (int n8 = 0; n8 < 16; ++n8) {
            v8h xv = *(const v8h*)(row + n8 * 8);
#pragma unroll
            for (int j = 0; j < 8; ++j) {
              int n = n8 * 8 + j;
              s += fast_tanh((float)xv[j] + Us[b][n]) * W2s[n];
            }
          }
          betas[b][tp] = s + b2v;
        }
      }
    }
    __syncthreads();

    // --- Phase C: ctx[b] = sum_t' beta[b][t'] * X[b][t'][:]  (per-b GEMV)
    {
      int b  = 2 * w + hi;        // each wave owns 2 batch rows
      int e0 = nn * 8;            // lanes 0..15 cover e = 0..127 contiguously
      const _Float16* xr = Xh + (size_t)(bg0 + b) * (TM1 * HE) + e0;
      float cacc[8] = {0, 0, 0, 0, 0, 0, 0, 0};
      for (int tp = 0; tp < TM1; ++tp) {
        float s = betas[b][tp];
        v8h xv = *(const v8h*)(xr + (size_t)tp * HE);
#pragma unroll
        for (int j = 0; j < 8; ++j) cacc[j] += s * (float)xv[j];
      }
#pragma unroll
      for (int j = 0; j < 8; ++j) ctxs[b][e0 + j] = cacc[j];
    }
    __syncthreads();

    // --- Phase D: y_tilde[b] = [ctx, y_t] @ Wfc + bfc
    if (tid < 16) {
      float s = bfcv + y_prev[(size_t)(bg0 + tid) * TM1 + t] * Wfcs[128];
      for (int e = 0; e < HE; ++e) s += ctxs[tid][e] * Wfcs[e];
      yts[tid] = s;
    }
    __syncthreads();

    // --- Phase E: Z = d @ Wh (16x512) + y_tilde*Wx + bl, then LSTM gates.
    // Wave w computes gate N-tiles {w, 8+w, 16+w, 24+w} = columns j,128+j,
    // 256+j,384+j for j in [16w,16w+16) -> whole gate update stays in regs.
    {
      v16h a[4];
#pragma unroll
      for (int kf = 0; kf < 4; ++kf) a[kf] = load_afrag(&DCh[nn][0], kf * 32 + hi * 16);
      v8f zi = {}, zf = {}, zg = {}, zo = {};
#pragma unroll
      for (int kf = 0; kf < 4; ++kf) {
        int ko = kf * 32 + hi * 16;
        v16h bi  = *(const v16h*)(WhT + (size_t)(ncol)       * 128 + ko);
        v16h bff = *(const v16h*)(WhT + (size_t)(128 + ncol) * 128 + ko);
        v16h bgg = *(const v16h*)(WhT + (size_t)(256 + ncol) * 128 + ko);
        v16h boo = *(const v16h*)(WhT + (size_t)(384 + ncol) * 128 + ko);
        zi = __builtin_amdgcn_wmma_f32_16x16x32_f16(false, a[kf], false, bi,  (short)0, zi, false, false);
        zf = __builtin_amdgcn_wmma_f32_16x16x32_f16(false, a[kf], false, bff, (short)0, zf, false, false);
        zg = __builtin_amdgcn_wmma_f32_16x16x32_f16(false, a[kf], false, bgg, (short)0, zg, false, false);
        zo = __builtin_amdgcn_wmma_f32_16x16x32_f16(false, a[kf], false, boo, (short)0, zo, false, false);
      }
      // all waves must finish reading DCh (A-frags span all 128 d-columns)
      // before anyone rewrites it below
      __syncthreads();
#pragma unroll
      for (int r = 0; r < 8; ++r) {
        int m = r + hi * 8;
        float yv = yts[m];
        float vi = fast_sigm(zi[r] + yv * wxi + bli);
        float vf = fast_sigm(zf[r] + yv * wxf + blf);
        float vg = fast_tanh(zg[r] + yv * wxg + blg);
        float vo = fast_sigm(zo[r] + yv * wxo + blo);
        float cn = vf * Cs[m][ncol] + vi * vg;
        float dn = vo * fast_tanh(cn);
        Cs[m][ncol] = cn;
        Ds[m][ncol] = dn;
        DCh[m][ncol] = (_Float16)dn;
        DCh[m][128 + ncol] = (_Float16)cn;
      }
    }
    __syncthreads();
  }

  // --- epilogue: out[b] = [d, ctx] @ Wf + bf
  if (tid < 16) {
    float s = bf[0];
    for (int e = 0; e < 128; ++e)
      s += Ds[tid][e] * Wf[e] + ctxs[tid][e] * Wf[128 + e];
    out[bg0 + tid] = s;
  }
}

// ---------------------------------------------------------------------------
extern "C" void kernel_launch(void* const* d_in, const int* in_sizes, int n_in,
                              void* d_out, int out_size, void* d_ws, size_t ws_size,
                              hipStream_t stream) {
  const float* X      = (const float*)d_in[0];   // (1024,127,128)
  const float* y_prev = (const float*)d_in[1];   // (1024,127)
  const float* W1     = (const float*)d_in[2];   // (384,128)
  const float* b1     = (const float*)d_in[3];   // (128)
  const float* W2     = (const float*)d_in[4];   // (128,1)
  const float* b2     = (const float*)d_in[5];   // (1)
  const float* Wfc    = (const float*)d_in[6];   // (129,1)
  const float* bfc    = (const float*)d_in[7];   // (1)
  const float* Wx     = (const float*)d_in[8];   // (1,512)
  const float* Wh     = (const float*)d_in[9];   // (128,512)
  const float* bl     = (const float*)d_in[10];  // (512)
  const float* Wf     = (const float*)d_in[11];  // (256,1)
  const float* bf     = (const float*)d_in[12];  // (1)
  float* out = (float*)d_out;

  // workspace carve-up (f16 elements)
  const size_t XB = (size_t)BSZ * TM1 * HE;  // 16,646,144
  _Float16* Xh    = (_Float16*)d_ws;
  _Float16* preXh = Xh + XB;
  _Float16* WhT   = preXh + XB;      // 512*128
  _Float16* W1dcT = WhT + 512 * 128; // 128*256
  _Float16* W1xT  = W1dcT + 128 * 256;

  prep_weights<<<448, 256, 0, stream>>>(W1, Wh, W1dcT, W1xT, WhT);
  conv_x<<<(int)((XB + 255) / 256), 256, 0, stream>>>(X, Xh, (int)XB);
  prex_gemm<<<(BSZ * TM1 / 16) / 8, 256, 0, stream>>>(Xh, W1xT, b1, preXh);
  decoder_main<<<BSZ / 16, 256, 0, stream>>>(X, y_prev, W2, b2, Wfc, bfc, Wx, bl,
                                             Wf, bf, Xh, preXh, W1dcT, WhT, out);
}
